// CGCNNConv_7275674599845
// MI455X (gfx1250) — matrix operand order repaired and checked by reading, not compiled, not run on hardware
//
#include <hip/hip_runtime.h>
#include <cstdint>
#include <cstddef>

#define HD 128

typedef __attribute__((ext_vector_type(16))) __bf16 v16bf;
typedef __attribute__((ext_vector_type(8)))  float  v8f;

union AFrag { v16bf v; uint4 q[2]; __bf16 h[16]; };

__device__ __forceinline__ void cvt8(const float* __restrict__ p, __bf16* d) {
  float4 a = *reinterpret_cast<const float4*>(p);
  float4 b = *reinterpret_cast<const float4*>(p + 4);
  d[0] = (__bf16)a.x; d[1] = (__bf16)a.y; d[2] = (__bf16)a.z; d[3] = (__bf16)a.w;
  d[4] = (__bf16)b.x; d[5] = (__bf16)b.y; d[6] = (__bf16)b.z; d[7] = (__bf16)b.w;
}

__device__ __forceinline__ float softplus_f(float x) {
  return x > 20.f ? x : log1pf(__expf(x));
}

__device__ __forceinline__ v8f wmma_bf16(v16bf a, v16bf b, v8f c) {
  return __builtin_amdgcn_wmma_f32_16x16x32_bf16(false, a, false, b, (short)0, c, false, false);
}

// Stage a row-major f32 weight matrix (K x 128) into LDS as bf16, permuted so
// each B-fragment (32x16 tile, per the CDNA5 WMMA B layout) is one contiguous
// 32B run per lane: block (c,nt) at offset (c*8+nt)*512, lane slot lane*16,
// element e holds W[c*32 + (e>>3)*16 + (lane>>4)*8 + (e&7)][nt*16 + (lane&15)].
__device__ __forceinline__ void stageW(const float* __restrict__ W, __bf16* sw,
                                       int nblk, int tid) {
  for (int i = tid; i < nblk * 512; i += 256) {
    int blk = i >> 9, rem = i & 511;
    int c = blk >> 3, nt = blk & 7;
    int l = rem >> 4, e = rem & 15;
    int k = c * 32 + ((e >> 3) << 4) + ((l >> 4) << 3) + (e & 7);
    int n = (nt << 4) + (l & 15);
    sw[i] = (__bf16)W[k * HD + n];
  }
}

__device__ __forceinline__ v16bf loadB(const __bf16* sw, int c, int nt, int lane) {
  AFrag B;
  const __bf16* p = sw + (((c << 3) + nt) << 9) + (lane << 4);
  B.q[0] = *reinterpret_cast<const uint4*>(p);
  B.q[1] = *reinterpret_cast<const uint4*>(p + 8);
  return B.v;
}

__global__ void zero_kernel(float* __restrict__ p, int n) {
  int i = blockIdx.x * 256 + threadIdx.x;
  if (i < n) p[i] = 0.f;
}

// Edge message kernel: 256 edges per workgroup, 8 waves, each wave does two
// M-tiles of 16 edges across all 128 output columns.
__global__ __launch_bounds__(256) void edge_mp_kernel(
    const float* __restrict__ nf, const int* __restrict__ ei,
    const float* __restrict__ ea,
    const float* __restrict__ W1, const float* __restrict__ b1,
    const float* __restrict__ W2, const float* __restrict__ b2,
    float* __restrict__ agg, int E) {
  extern __shared__ char smem[];
  __bf16* sW1  = reinterpret_cast<__bf16*>(smem); // 96 blocks * 512
  __bf16* sW2  = sW1 + 96 * 512;                  // 32 blocks * 512
  __bf16* sScr = sW2 + 32 * 512;                  // 8 waves * 512 (16x32 bf16)

  const int tid  = threadIdx.x;
  const int lane = tid & 31;
  const int wave = tid >> 5;
  const int hh   = lane >> 4;   // lane half
  const int ln   = lane & 15;
  const int* ei0 = ei;          // row (scatter target + first gather)
  const int* ei1 = ei + E;      // col

  stageW(W1, sW1, 96, tid);     // K=384 -> 12 chunks x 8 ntiles
  stageW(W2, sW2, 32, tid);     // K=128 -> 4  chunks x 8 ntiles
  __syncthreads();

  __bf16* scr = sScr + wave * 512;

  for (int mt = 0; mt < 2; ++mt) {
    const int ebase = blockIdx.x * 256 + ((mt << 3) + wave) * 16;
    const int eL = ebase + ln;
    const int eC = eL < E ? eL : E - 1;
    const int r0 = ei0[eC], c0 = ei1[eC];
    const float* srcR = nf + (size_t)r0 * HD;
    const float* srcC = nf + (size_t)c0 * HD;
    const float* srcE = ea + (size_t)eC * HD;

    // ---- layer 1: [x_row | x_col | e_attr] @ W1 + b1 ----
    v8f acc[8];
#pragma unroll
    for (int nt = 0; nt < 8; ++nt) {
      float bv = b1[(nt << 4) + ln];
#pragma unroll
      for (int r = 0; r < 8; ++r) acc[nt][r] = bv;
    }

    for (int c = 0; c < 12; ++c) {
      const float* s = (c < 4) ? (srcR + (c << 5))
                               : (c < 8 ? (srcC + ((c - 4) << 5))
                                        : (srcE + ((c - 8) << 5)));
      AFrag A;  // CDNA5 A layout: elems 0..7 = K hh*8.., elems 8..15 = K 16+hh*8..
      cvt8(s + hh * 8, A.h);
      cvt8(s + 16 + hh * 8, A.h + 8);
#pragma unroll
      for (int nt = 0; nt < 8; ++nt)
        acc[nt] = wmma_bf16(A.v, loadB(sW1, c, nt, lane), acc[nt]);
    }

    // ---- softplus + layer 2: hidden @ W2 + b2 ----
    v8f acc2[8];
#pragma unroll
    for (int nt = 0; nt < 8; ++nt) {
      float bv = b2[(nt << 4) + ln];
#pragma unroll
      for (int r = 0; r < 8; ++r) acc2[nt][r] = bv;
    }

#pragma unroll
    for (int c2 = 0; c2 < 4; ++c2) {
      // bounce C-fragments 2*c2, 2*c2+1 through wave-private LDS to rebuild
      // the A layout (same-wave DS ordering, no barrier needed)
#pragma unroll
      for (int p = 0; p < 2; ++p) {
        const int nt = (c2 << 1) + p;
#pragma unroll
        for (int r = 0; r < 8; ++r) {
          const int m = r + (hh << 3);
          scr[m * 32 + (p << 4) + ln] = (__bf16)softplus_f(acc[nt][r]);
        }
      }
      AFrag A2;
      const __bf16* sm = scr + ln * 32;
      A2.q[0] = *reinterpret_cast<const uint4*>(sm + hh * 8);
      A2.q[1] = *reinterpret_cast<const uint4*>(sm + 16 + hh * 8);
#pragma unroll
      for (int nt = 0; nt < 8; ++nt)
        acc2[nt] = wmma_bf16(A2.v, loadB(sW2, c2, nt, lane), acc2[nt]);
    }

    // ---- scatter-add messages to agg[row] ----
#pragma unroll
    for (int r = 0; r < 8; ++r) {
      const int m = r + (hh << 3);
      const int e = ebase + m;
      if (e < E) {
        const int rr = ei0[e];
        float* dst = agg + (size_t)rr * HD + ln;
#pragma unroll
        for (int nt = 0; nt < 8; ++nt)
          atomicAdd(dst + (nt << 4), acc2[nt][r]);
      }
    }
  }
}

// Node update kernel: 256 nodes per workgroup; also accumulates BN statistics.
__global__ __launch_bounds__(256) void node_up_kernel(
    const float* __restrict__ nf, const float* __restrict__ agg,
    const float* __restrict__ U1, const float* __restrict__ ub1,
    const float* __restrict__ U2, const float* __restrict__ ub2,
    float* __restrict__ out, float* __restrict__ sums,
    float* __restrict__ sumsq, int N) {
  extern __shared__ char smem[];
  __bf16* sU1  = reinterpret_cast<__bf16*>(smem); // 64 blocks * 512
  __bf16* sU2  = sU1 + 64 * 512;                  // 32 blocks * 512
  __bf16* sScr = sU2 + 32 * 512;                  // 8 waves * 512

  const int tid  = threadIdx.x;
  const int lane = tid & 31;
  const int wave = tid >> 5;
  const int hh   = lane >> 4;
  const int ln   = lane & 15;

  stageW(U1, sU1, 64, tid);  // K=256 -> 8 chunks x 8 ntiles
  stageW(U2, sU2, 32, tid);
  __syncthreads();

  __bf16* scr = sScr + wave * 512;

  for (int mt = 0; mt < 2; ++mt) {
    const int nbase = blockIdx.x * 256 + ((mt << 3) + wave) * 16;
    const int nd = nbase + ln;
    const int ndc = nd < N ? nd : N - 1;
    const float* srcA = nf  + (size_t)ndc * HD;
    const float* srcB = agg + (size_t)ndc * HD;

    v8f acc[8];
#pragma unroll
    for (int nt = 0; nt < 8; ++nt) {
      float bv = ub1[(nt << 4) + ln];
#pragma unroll
      for (int r = 0; r < 8; ++r) acc[nt][r] = bv;
    }

    for (int c = 0; c < 8; ++c) {
      const float* s = (c < 4) ? (srcA + (c << 5)) : (srcB + ((c - 4) << 5));
      AFrag A;
      cvt8(s + hh * 8, A.h);
      cvt8(s + 16 + hh * 8, A.h + 8);
#pragma unroll
      for (int nt = 0; nt < 8; ++nt)
        acc[nt] = wmma_bf16(A.v, loadB(sU1, c, nt, lane), acc[nt]);
    }

    v8f acc2[8];
#pragma unroll
    for (int nt = 0; nt < 8; ++nt) {
      float bv = ub2[(nt << 4) + ln];
#pragma unroll
      for (int r = 0; r < 8; ++r) acc2[nt][r] = bv;
    }

#pragma unroll
    for (int c2 = 0; c2 < 4; ++c2) {
#pragma unroll
      for (int p = 0; p < 2; ++p) {
        const int nt = (c2 << 1) + p;
#pragma unroll
        for (int r = 0; r < 8; ++r) {
          const int m = r + (hh << 3);
          scr[m * 32 + (p << 4) + ln] = (__bf16)softplus_f(acc[nt][r]);
        }
      }
      AFrag A2;
      const __bf16* sm = scr + ln * 32;
      A2.q[0] = *reinterpret_cast<const uint4*>(sm + hh * 8);
      A2.q[1] = *reinterpret_cast<const uint4*>(sm + 16 + hh * 8);
#pragma unroll
      for (int nt = 0; nt < 8; ++nt)
        acc2[nt] = wmma_bf16(A2.v, loadB(sU2, c2, nt, lane), acc2[nt]);
    }

    // store pre-BN output + accumulate batch statistics
#pragma unroll
    for (int nt = 0; nt < 8; ++nt) {
      float s = 0.f, s2 = 0.f;
#pragma unroll
      for (int r = 0; r < 8; ++r) {
        const int m = r + (hh << 3);
        const int node = nbase + m;
        if (node < N) {
          const float v = acc2[nt][r];
          out[(size_t)node * HD + (nt << 4) + ln] = v;
          s += v;
          s2 += v * v;
        }
      }
      atomicAdd(&sums[(nt << 4) + ln], s);
      atomicAdd(&sumsq[(nt << 4) + ln], s2);
    }
  }
}

__global__ void bn_kernel(float* __restrict__ out,
                          const float* __restrict__ sums,
                          const float* __restrict__ sumsq,
                          const float* __restrict__ gamma,
                          const float* __restrict__ beta,
                          int N, size_t total) {
  size_t i = (size_t)blockIdx.x * 256 + threadIdx.x;
  if (i < total) {
    const int col = (int)(i & (HD - 1));
    const float invN = 1.f / (float)N;
    const float mu  = sums[col] * invN;
    const float var = sumsq[col] * invN - mu * mu;
    out[i] = (out[i] - mu) * rsqrtf(var + 1e-5f) * gamma[col] + beta[col];
  }
}

extern "C" void kernel_launch(void* const* d_in, const int* in_sizes, int n_in,
                              void* d_out, int out_size, void* d_ws, size_t ws_size,
                              hipStream_t stream) {
  (void)n_in; (void)out_size; (void)ws_size;
  const float* nf    = (const float*)d_in[0];
  const int*   ei    = (const int*)  d_in[1];
  const float* ea    = (const float*)d_in[2];
  const float* W1    = (const float*)d_in[3];
  const float* b1    = (const float*)d_in[4];
  const float* W2    = (const float*)d_in[5];
  const float* b2    = (const float*)d_in[6];
  const float* U1    = (const float*)d_in[7];
  const float* ub1   = (const float*)d_in[8];
  const float* U2    = (const float*)d_in[9];
  const float* ub2   = (const float*)d_in[10];
  const float* gamma = (const float*)d_in[11];
  const float* beta  = (const float*)d_in[12];

  const int N = in_sizes[0] / HD;
  const int E = in_sizes[1] / 2;

  float* agg   = (float*)d_ws;            // N*128
  float* sums  = agg + (size_t)N * HD;    // 128
  float* sumsq = sums + HD;               // 128

  {
    const int tot = N * HD + 2 * HD;
    zero_kernel<<<(tot + 255) / 256, 256, 0, stream>>>(agg, tot);
  }

  const size_t edge_smem = (size_t)(96 * 512 + 32 * 512 + 8 * 512) * sizeof(__bf16);
  edge_mp_kernel<<<(E + 255) / 256, 256, edge_smem, stream>>>(
      nf, ei, ea, W1, b1, W2, b2, agg, E);

  const size_t node_smem = (size_t)(64 * 512 + 32 * 512 + 8 * 512) * sizeof(__bf16);
  node_up_kernel<<<(N + 255) / 256, 256, node_smem, stream>>>(
      nf, agg, U1, ub1, U2, ub2, (float*)d_out, sums, sumsq, N);

  {
    const size_t tot = (size_t)N * HD;
    bn_kernel<<<(tot + 255) / 256, 256, 0, stream>>>(
        (float*)d_out, sums, sumsq, gamma, beta, N, tot);
  }
}
